// CrossAttention_29815662969497
// MI455X (gfx1250) — matrix operand order repaired and checked by reading, not compiled, not run on hardware
//
#include <hip/hip_runtime.h>
#include <cstdint>
#include <cstddef>

// Problem constants (from reference)
#define BATCH  2
#define NSEQ   2048
#define DMODEL 512
#define NH     8
#define DHEAD  64
#define PMAX   64   // rel-pos clamp

typedef __attribute__((ext_vector_type(16))) __bf16 v16bf;
typedef __attribute__((ext_vector_type(8)))  __bf16 v8bf;
typedef __attribute__((ext_vector_type(8)))  float  v8f;

// ---------------------------------------------------------------- helpers
static __device__ __forceinline__ __bf16 f2bf(float f) {
  unsigned u = __builtin_bit_cast(unsigned, f);
  u += 0x7FFFu + ((u >> 16) & 1u);             // round-to-nearest-even
  unsigned short s = (unsigned short)(u >> 16);
  return __builtin_bit_cast(__bf16, s);
}

static __device__ __forceinline__ v8f wmma_bf16(v16bf a, v16bf b, v8f c) {
  // D = A(16x32 bf16) * B(32x16 bf16) + C(16x16 f32)
  return __builtin_amdgcn_wmma_f32_16x16x32_bf16(
      /*neg_a=*/false, a, /*neg_b=*/false, b,
      /*c_mod=*/(short)0, c, /*reuse_a=*/false, /*reuse_b=*/false);
}

// A-fragment: two contiguous 8-element (16B) chunks per lane
static __device__ __forceinline__ v16bf load_a16(const __bf16* lo, const __bf16* hi) {
  v8bf l = *(const v8bf*)lo;
  v8bf h = *(const v8bf*)hi;
  v16bf r;
#pragma unroll
  for (int t = 0; t < 8; ++t) { r[t] = l[t]; r[t + 8] = h[t]; }
  return r;
}

static __device__ __forceinline__ float rmax16(float x) {
  x = fmaxf(x, __shfl_xor(x, 1));
  x = fmaxf(x, __shfl_xor(x, 2));
  x = fmaxf(x, __shfl_xor(x, 4));
  x = fmaxf(x, __shfl_xor(x, 8));
  return x;   // max across the 16-lane group (row of a C-layout tile)
}
static __device__ __forceinline__ float rsum16(float x) {
  x += __shfl_xor(x, 1);
  x += __shfl_xor(x, 2);
  x += __shfl_xor(x, 4);
  x += __shfl_xor(x, 8);
  return x;
}

static __device__ __forceinline__ int clamp_rel(int d) {
  d = d < -PMAX ? -PMAX : d;
  d = d >  PMAX ?  PMAX : d;
  return d + PMAX;   // [0, 128]
}

// ---------------------------------------------------------------- converts
__global__ void __launch_bounds__(256) k_cvt_bf16(const float* __restrict__ src,
                                                  __bf16* __restrict__ dst, int n) {
  int t = blockIdx.x * 256 + threadIdx.x;
  if (t < n) dst[t] = f2bf(src[t]);
}

// dst[n*K + k] = src[k*Ncol + n]   (transpose + convert, weights are tiny)
__global__ void __launch_bounds__(256) k_cvt_transpose(const float* __restrict__ src,
                                                       __bf16* __restrict__ dst,
                                                       int K, int Ncol) {
  int t = blockIdx.x * 256 + threadIdx.x;
  if (t < K * Ncol) {
    int n = t / K;
    int k = t - n * K;
    dst[t] = f2bf(src[k * Ncol + n]);
  }
}

// ---------------------------------------------------------------- GEMM
// C[M,512] = A[M,512](bf16) @ W (W given transposed bf16: Wt[n][k]).
// One wave computes a 16x64 tile: 1 A-frag reused across 4 B-frags per k-step.
// mode 0: store bf16 identity layout (Q,K)
// mode 1: store bf16 transposed into Vt[b][h][d][j]
// mode 2: store f32 with bias (final projection into d_out)
__global__ void __launch_bounds__(256) k_gemm_bf16(const __bf16* __restrict__ A,
                                                   const __bf16* __restrict__ Wt,
                                                   void* __restrict__ out,
                                                   const float* __restrict__ bias,
                                                   int mode) {
  const int lane = threadIdx.x & 31;
  const int wave = threadIdx.x >> 5;
  const int r    = lane & 15;
  const int half = lane >> 4;

  const int i0 = (blockIdx.x * 8 + wave) * 16;   // 32 blocks * 8 waves * 16 = 4096 rows
  const int n0 = blockIdx.y * 64;                // 8 blocks * 64 = 512 cols

  v8f acc[4];
#pragma unroll
  for (int c = 0; c < 4; ++c)
#pragma unroll
    for (int v = 0; v < 8; ++v) acc[c][v] = 0.0f;

  const __bf16* arow = A + (size_t)(i0 + r) * DMODEL;

  for (int kk = 0; kk < DMODEL; kk += 32) {
    v16bf af = load_a16(arow + kk + 16 * half, arow + kk + 16 * half + 8);
    __builtin_prefetch(arow + kk + 64, 0, 1);
#pragma unroll
    for (int c = 0; c < 4; ++c) {
      const __bf16* bp = Wt + (size_t)(n0 + c * 16 + r) * DMODEL + kk + 16 * half;
      v16bf bf_ = *(const v16bf*)bp;
      acc[c] = wmma_bf16(af, bf_, acc[c]);
    }
  }

  if (mode == 0) {
    __bf16* o = (__bf16*)out;
#pragma unroll
    for (int c = 0; c < 4; ++c)
#pragma unroll
      for (int v = 0; v < 8; ++v)
        o[(size_t)(i0 + v + 8 * half) * DMODEL + n0 + c * 16 + r] = f2bf(acc[c][v]);
  } else if (mode == 1) {
    __bf16* o = (__bf16*)out;   // Vt[((b*8+h)*64+d)*2048 + j]
#pragma unroll
    for (int c = 0; c < 4; ++c) {
      int colg = n0 + c * 16 + r;
      int h = colg >> 6, d = colg & 63;
#pragma unroll
      for (int v = 0; v < 8; ++v) {
        int ig = i0 + v + 8 * half;
        int b  = ig >> 11, j = ig & (NSEQ - 1);
        o[((size_t)((b * NH + h) * DHEAD + d)) * NSEQ + j] = f2bf(acc[c][v]);
      }
    }
  } else {
    float* o = (float*)out;
#pragma unroll
    for (int c = 0; c < 4; ++c) {
      int colg = n0 + c * 16 + r;
      float bv = bias ? bias[colg] : 0.0f;
#pragma unroll
      for (int v = 0; v < 8; ++v)
        o[(size_t)(i0 + v + 8 * half) * DMODEL + colg] = acc[c][v] + bv;
    }
  }
}

// ---------------------------------------------------------------- attention
// grid.x = B*H (16), grid.y = 16; block = 256 (8 waves). One wave owns a
// 16-row i-tile and streams all 2048 keys in 32-column j-blocks with an
// online (flash) softmax. S tiles are re-laid-out through a per-wave LDS
// slab so the probability tile can feed the PV WMMA as an A-operand.
__global__ void __launch_bounds__(256) k_attention(const __bf16* __restrict__ Qb,
                                                   const __bf16* __restrict__ Kb,
                                                   const __bf16* __restrict__ Vt,
                                                   const float* __restrict__ rel_pos,
                                                   const float* __restrict__ c_emb,
                                                   __bf16* __restrict__ ctx) {
  const int lane = threadIdx.x & 31;
  const int wave = threadIdx.x >> 5;
  const int col  = lane & 15;
  const int half = lane >> 4;

  const int bh = blockIdx.x;
  const int b  = bh >> 3;
  const int h  = bh & 7;
  const int i0 = (blockIdx.y * 8 + wave) * 16;

  __shared__ float s_rel[2 * PMAX + 1];
  __shared__ float s_gate[2 * PMAX + 1];
  __shared__ __align__(16) __bf16 s_p[8][16][40];   // per-wave 16x32 tile, padded rows

  for (int t = threadIdx.x; t < 2 * PMAX + 1; t += 256) {
    s_rel[t]  = rel_pos[t * NH + h];
    s_gate[t] = c_emb[t * NH + h];
  }
  __syncthreads();

  // Q fragments for this wave's 16 rows (k = d, 64 = two 32-wide steps)
  const __bf16* qrow = Qb + ((size_t)((b * NSEQ + i0 + col) * NH) + h) * DHEAD;
  v16bf aq0 = load_a16(qrow + 16 * half,      qrow + 16 * half + 8);
  v16bf aq1 = load_a16(qrow + 32 + 16 * half, qrow + 40 + 16 * half);

  v8f o[4];
  float m[8], l[8];
#pragma unroll
  for (int c = 0; c < 4; ++c)
#pragma unroll
    for (int v = 0; v < 8; ++v) o[c][v] = 0.0f;
#pragma unroll
  for (int v = 0; v < 8; ++v) { m[v] = -__builtin_inff(); l[v] = 0.0f; }

  const float scale = 0.125f;   // DH^-0.5
  __bf16* slab = &s_p[wave][0][0];

  for (int j0 = 0; j0 < NSEQ; j0 += 32) {
    // ---- S = Q K^T for two 16-col tiles
    v8f sacc[2];
#pragma unroll
    for (int jh = 0; jh < 2; ++jh) {
      const __bf16* kp =
          Kb + ((size_t)((b * NSEQ + j0 + 16 * jh + col) * NH) + h) * DHEAD + 16 * half;
      v16bf bk0 = *(const v16bf*)kp;          // d 0..31 slice
      v16bf bk1 = *(const v16bf*)(kp + 32);   // d 32..63 slice
      v8f z;
#pragma unroll
      for (int v = 0; v < 8; ++v) z[v] = 0.0f;
      z = wmma_bf16(aq0, bk0, z);
      z = wmma_bf16(aq1, bk1, z);
      sacc[jh] = z;
    }

    // ---- bias + online softmax + gate; write P tile (bf16) to LDS
    const int jA = j0 + col, jB = j0 + 16 + col;
#pragma unroll
    for (int v = 0; v < 8; ++v) {
      int irow = i0 + v + 8 * half;
      int dA = clamp_rel(jA - irow);
      int dB = clamp_rel(jB - irow);
      float sA = (sacc[0][v] + s_rel[dA]) * scale;
      float sB = (sacc[1][v] + s_rel[dB]) * scale;

      float tmax = rmax16(fmaxf(sA, sB));
      float mnew = fmaxf(m[v], tmax);
      float al   = __expf(m[v] - mnew);
      float pA   = __expf(sA - mnew);
      float pB   = __expf(sB - mnew);
      l[v] = l[v] * al + rsum16(pA + pB);
      m[v] = mnew;
      o[0][v] *= al; o[1][v] *= al; o[2][v] *= al; o[3][v] *= al;

      __bf16* srow = slab + (v + 8 * half) * 40;
      srow[col]      = f2bf(pA * s_gate[dA]);
      srow[16 + col] = f2bf(pB * s_gate[dB]);
    }

    // in-wave LDS RAW: wait for DS stores, then read P as an A-fragment
    asm volatile("s_wait_dscnt 0" ::: "memory");
    const __bf16* rp = slab + col * 40;
    v16bf pa = load_a16(rp + 16 * half, rp + 16 * half + 8);

    // ---- O += P @ V  (k = j, 32 wide; 4 d-chunks of 16)
#pragma unroll
    for (int c = 0; c < 4; ++c) {
      const __bf16* vp =
          Vt + ((size_t)((b * NH + h) * DHEAD + c * 16 + col)) * NSEQ + j0 + 16 * half;
      v16bf bv = *(const v16bf*)vp;
      o[c] = wmma_bf16(pa, bv, o[c]);
    }
  }

  // ---- normalize and emit bf16 context in [b, i, h*64+d] layout
  float inv[8];
#pragma unroll
  for (int v = 0; v < 8; ++v) inv[v] = 1.0f / l[v];
#pragma unroll
  for (int c = 0; c < 4; ++c)
#pragma unroll
    for (int v = 0; v < 8; ++v) {
      int ig = i0 + v + 8 * half;
      ctx[(size_t)(b * NSEQ + ig) * DMODEL + h * DHEAD + c * 16 + col] =
          f2bf(o[c][v] * inv[v]);
    }
}

// ---------------------------------------------------------------- launch
extern "C" void kernel_launch(void* const* d_in, const int* in_sizes, int n_in,
                              void* d_out, int out_size, void* d_ws, size_t ws_size,
                              hipStream_t stream) {
  (void)in_sizes; (void)n_in; (void)out_size; (void)ws_size;

  const float* x    = (const float*)d_in[0];
  const float* Wq   = (const float*)d_in[1];
  const float* Wk   = (const float*)d_in[2];
  const float* Wv   = (const float*)d_in[3];
  const float* relp = (const float*)d_in[4];
  const float* cemb = (const float*)d_in[5];
  const float* Wo   = (const float*)d_in[6];
  const float* bo   = (const float*)d_in[7];

  char* ws = (char*)d_ws;
  const size_t SZ_X  = (size_t)BATCH * NSEQ * DMODEL * 2;   // 4 MB bf16
  const size_t SZ_W  = (size_t)DMODEL * DMODEL * 2;         // 0.5 MB bf16
  __bf16* Xb  = (__bf16*)(ws);
  __bf16* Wtq = (__bf16*)(ws + SZ_X);
  __bf16* Wtk = (__bf16*)(ws + SZ_X + SZ_W);
  __bf16* Wtv = (__bf16*)(ws + SZ_X + 2 * SZ_W);
  __bf16* Wto = (__bf16*)(ws + SZ_X + 3 * SZ_W);
  __bf16* Qb  = (__bf16*)(ws + SZ_X + 4 * SZ_W);
  __bf16* Kb  = (__bf16*)(ws + 2 * SZ_X + 4 * SZ_W);
  __bf16* Vt  = (__bf16*)(ws + 3 * SZ_X + 4 * SZ_W);
  __bf16* Ctx = (__bf16*)(ws + 4 * SZ_X + 4 * SZ_W);

  const int nX = BATCH * NSEQ * DMODEL;           // 2,097,152
  const int nW = DMODEL * DMODEL;                 // 262,144

  k_cvt_bf16<<<(nX + 255) / 256, 256, 0, stream>>>(x, Xb, nX);
  k_cvt_transpose<<<(nW + 255) / 256, 256, 0, stream>>>(Wq, Wtq, DMODEL, DMODEL);
  k_cvt_transpose<<<(nW + 255) / 256, 256, 0, stream>>>(Wk, Wtk, DMODEL, DMODEL);
  k_cvt_transpose<<<(nW + 255) / 256, 256, 0, stream>>>(Wv, Wtv, DMODEL, DMODEL);
  k_cvt_transpose<<<(nW + 255) / 256, 256, 0, stream>>>(Wo, Wto, DMODEL, DMODEL);

  dim3 gGemm(32, 8);   // 4096/(16*8) x 512/64
  k_gemm_bf16<<<gGemm, 256, 0, stream>>>(Xb, Wtq, (void*)Qb, nullptr, 0);
  k_gemm_bf16<<<gGemm, 256, 0, stream>>>(Xb, Wtk, (void*)Kb, nullptr, 0);
  k_gemm_bf16<<<gGemm, 256, 0, stream>>>(Xb, Wtv, (void*)Vt, nullptr, 1);

  dim3 gAttn(BATCH * NH, NSEQ / (16 * 8));   // 16 x 16
  k_attention<<<gAttn, 256, 0, stream>>>(Qb, Kb, Vt, relp, cemb, Ctx);

  k_gemm_bf16<<<gGemm, 256, 0, stream>>>(Ctx, Wto, d_out, bo, 2);
}
